// MultiHeadAttention_11622181503577
// MI455X (gfx1250) — compile-verified, matrix operand order checked by reading, not run on hardware
//
#include <hip/hip_runtime.h>

// ---------------- types ----------------
typedef __attribute__((ext_vector_type(16))) __bf16 bf16x16;
typedef __attribute__((ext_vector_type(8)))  __bf16 bf16x8;
typedef __attribute__((ext_vector_type(8)))  float  f32x8;
typedef int v4i_ __attribute__((vector_size(16)));

#define B_NUM   4
#define S_LEN   2048
#define D_DIM   1024
#define H_NUM   16
#define DK_DIM  64
#define NEGF    (-3.0e38f)

// CDNA5 async global->LDS direct copy (ASYNCcnt-tracked), with fallback.
#if defined(__gfx1250__) && __has_builtin(__builtin_amdgcn_global_load_async_to_lds_b128)
#define HAVE_ASYNC_LDS 1
#else
#define HAVE_ASYNC_LDS 0
#endif

#if HAVE_ASYNC_LDS
#define AS1 __attribute__((address_space(1)))
#define AS3 __attribute__((address_space(3)))
__device__ __forceinline__ void async_copy_b128(const void* g, void* l) {
  __builtin_amdgcn_global_load_async_to_lds_b128((AS1 v4i_*)g, (AS3 v4i_*)l, 0, 0);
}
__device__ __forceinline__ void wait_async0() {
#if __has_builtin(__builtin_amdgcn_s_wait_asynccnt)
  __builtin_amdgcn_s_wait_asynccnt(0);
#else
  asm volatile("s_wait_asynccnt 0x0" ::: "memory");
#endif
}
#endif

__device__ __forceinline__ bf16x16 join8(bf16x8 lo, bf16x8 hi) {
  return __builtin_shufflevector(lo, hi, 0,1,2,3,4,5,6,7,8,9,10,11,12,13,14,15);
}

__device__ __forceinline__ f32x8 wmma_bf16(bf16x16 a, bf16x16 b, f32x8 c) {
  // D = A(16x32 bf16) * B(32x16 bf16) + C(16x16 f32)
  return __builtin_amdgcn_wmma_f32_16x16x32_bf16(false, a, false, b, (short)0, c,
                                                 false, false);
}

// ---------------- kernel 1: weight repack fp32 -> bf16 ----------------
__global__ void mha_cvt_kernel(const float* __restrict__ Wq,
                               const float* __restrict__ Wkv,
                               const float* __restrict__ bq,
                               const float* __restrict__ bkv,
                               const float* __restrict__ Wo,
                               __bf16* __restrict__ Wqkvb,
                               __bf16* __restrict__ Wob,
                               float*  __restrict__ biasq) {
  int idx = blockIdx.x * 256 + threadIdx.x;
  if (idx < 1024 * 3072) {
    int row = idx / 3072, col = idx - row * 3072;
    float v = (col < 1024) ? Wq[row * 1024 + col] : Wkv[row * 2048 + (col - 1024)];
    Wqkvb[idx] = (__bf16)v;
  }
  if (idx < 1024 * 1024) Wob[idx] = (__bf16)Wo[idx];
  if (idx < 3072) biasq[idx] = (idx < 1024) ? bq[idx] : bkv[idx - 1024];
}

// ---------------- kernel 2: LayerNorm (one wave32 per row) ----------------
__global__ __launch_bounds__(256) void mha_ln_kernel(const float* __restrict__ x,
                                                     const float* __restrict__ gamma,
                                                     const float* __restrict__ beta,
                                                     __bf16* __restrict__ xn) {
  const int lane = threadIdx.x & 31;
  const int wv   = threadIdx.x >> 5;
  const int row  = blockIdx.x * 8 + wv;
  const float* xp = x + (size_t)row * D_DIM;
  float r[32];
  float sum = 0.f;
#pragma unroll
  for (int i = 0; i < 32; ++i) { r[i] = xp[i * 32 + lane]; sum += r[i]; }
#pragma unroll
  for (int off = 1; off < 32; off <<= 1) sum += __shfl_xor(sum, off, 32);
  float mu = sum * (1.0f / 1024.0f);
  float vs = 0.f;
#pragma unroll
  for (int i = 0; i < 32; ++i) { float d = r[i] - mu; vs += d * d; }
#pragma unroll
  for (int off = 1; off < 32; off <<= 1) vs += __shfl_xor(vs, off, 32);
  float rs = rsqrtf(vs * (1.0f / 1024.0f) + 1e-6f);
  __bf16* op = xn + (size_t)row * D_DIM;
#pragma unroll
  for (int i = 0; i < 32; ++i) {
    int c = i * 32 + lane;
    op[c] = (__bf16)((r[i] - mu) * rs * gamma[c] + beta[c]);
  }
}

// ---------------- kernel 3/5: bf16 WMMA GEMM, 128x128x64 tiles ----------------
#define BM  128
#define BN  128
#define BKK 64
#define BKP 72   // padded K-pitch for transposed B tile (144B rows, 16B aligned)

__global__ __launch_bounds__(256) void mha_gemm_kernel(
    const __bf16* __restrict__ A,   // [M x K] row-major
    const __bf16* __restrict__ Bw,  // [K x N] row-major
    int N, int K,
    const float* __restrict__ bias,   // [N]
    const float* __restrict__ resid,  // epi 1: [M x N]
    float* __restrict__ outF,         // epi 1
    __bf16* __restrict__ qb, __bf16* __restrict__ kb, __bf16* __restrict__ vb, // epi 0
    int epi) {
  __shared__ __bf16 As[BM * BKK];
  __shared__ __bf16 Bs[BN * BKP];   // transposed: Bs[n][k]
  const int tid   = threadIdx.x;
  const int lane  = tid & 31;
  const int wv    = tid >> 5;
  const int waveM = wv >> 1;   // 0..3
  const int waveN = wv & 1;    // 0..1
  const int lrow  = lane & 15;
  const int half8 = (lane >> 4) << 3;  // 0 or 8 (16-bit A/B K-half select)
  const int mBase = blockIdx.x * BM;
  const int nBase = blockIdx.y * BN;

  f32x8 acc[2][4];
#pragma unroll
  for (int mt = 0; mt < 2; ++mt)
#pragma unroll
    for (int nt = 0; nt < 4; ++nt)
      acc[mt][nt] = f32x8{0.f,0.f,0.f,0.f,0.f,0.f,0.f,0.f};

  for (int k0 = 0; k0 < K; k0 += BKK) {
    __syncthreads();
    // ---- A tile [BM][BKK]: straight copy, async direct-to-LDS ----
#if HAVE_ASYNC_LDS
#pragma unroll
    for (int i = 0; i < 4; ++i) {
      int f = i * 2048 + tid * 8;
      int r = f >> 6, c = f & 63;
      async_copy_b128(&A[(size_t)(mBase + r) * K + k0 + c], &As[f]);
    }
#else
    {
      bf16x8 ta[4];
#pragma unroll
      for (int i = 0; i < 4; ++i) {
        int f = i * 2048 + tid * 8;
        int r = f >> 6, c = f & 63;
        ta[i] = *(const bf16x8*)&A[(size_t)(mBase + r) * K + k0 + c];
      }
#pragma unroll
      for (int i = 0; i < 4; ++i) *(bf16x8*)&As[i * 2048 + tid * 8] = ta[i];
    }
#endif
    // ---- B tile: load batched, store transposed into Bs[n][k] ----
    {
      bf16x8 tb[4];
#pragma unroll
      for (int i = 0; i < 4; ++i) {
        int f = i * 2048 + tid * 8;
        int r = f >> 7, c = f & 127;
        tb[i] = *(const bf16x8*)&Bw[(size_t)(k0 + r) * N + nBase + c];
      }
#pragma unroll
      for (int i = 0; i < 4; ++i) {
        int f = i * 2048 + tid * 8;
        int r = f >> 7, c = f & 127;
#pragma unroll
        for (int j = 0; j < 8; ++j) Bs[(c + j) * BKP + r] = tb[i][j];
      }
    }
    if (k0 + BKK < K) {  // global_prefetch_b8 of next tiles
      __builtin_prefetch(&A[(size_t)(mBase + (tid >> 1)) * K + k0 + BKK], 0, 0);
      __builtin_prefetch(&Bw[(size_t)(k0 + BKK + (tid >> 2)) * N + nBase], 0, 0);
    }
#if HAVE_ASYNC_LDS
    wait_async0();
#endif
    __syncthreads();
#pragma unroll
    for (int kk = 0; kk < BKK; kk += 32) {
      bf16x16 af[2];
#pragma unroll
      for (int mt = 0; mt < 2; ++mt) {
        const __bf16* rp = &As[(waveM * 32 + mt * 16 + lrow) * BKK + kk + half8];
        af[mt] = join8(*(const bf16x8*)rp, *(const bf16x8*)(rp + 16));
      }
      bf16x16 bfr[4];
#pragma unroll
      for (int nt = 0; nt < 4; ++nt) {
        const __bf16* cp = &Bs[(waveN * 64 + nt * 16 + lrow) * BKP + kk + half8];
        bfr[nt] = join8(*(const bf16x8*)cp, *(const bf16x8*)(cp + 16));
      }
#pragma unroll
      for (int mt = 0; mt < 2; ++mt)
#pragma unroll
        for (int nt = 0; nt < 4; ++nt)
          acc[mt][nt] = wmma_bf16(af[mt], bfr[nt], acc[mt][nt]);
    }
  }

  // epilogue
#pragma unroll
  for (int mt = 0; mt < 2; ++mt) {
#pragma unroll
    for (int nt = 0; nt < 4; ++nt) {
#pragma unroll
      for (int s = 0; s < 8; ++s) {
        int gRow = mBase + waveM * 32 + mt * 16 + s + half8;
        int gCol = nBase + waveN * 64 + nt * 16 + lrow;
        float vv = acc[mt][nt][s] + bias[gCol];
        if (epi == 0) {
          // scatter into q/k/v  [B,H,S,64] bf16
          int b = gRow >> 11, sIdx = gRow & 2047;
          int region = gCol >> 10;
          int cc = gCol & 1023;
          int h = cc >> 6, d = cc & 63;
          size_t idx = (((size_t)b * H_NUM + h) * S_LEN + sIdx) * DK_DIM + d;
          __bf16 bvv = (__bf16)vv;
          if (region == 0)      qb[idx] = bvv;
          else if (region == 1) kb[idx] = bvv;
          else                  vb[idx] = bvv;
        } else {
          size_t idx = (size_t)gRow * N + gCol;
          outF[idx] = vv + resid[idx];
        }
      }
    }
  }
}

// ---------------- kernel 4: flash attention, WMMA QK^T and PV ----------------
__global__ __launch_bounds__(256) void mha_attn_kernel(
    const __bf16* __restrict__ q, const __bf16* __restrict__ k,
    const __bf16* __restrict__ v, const int* __restrict__ lengths,
    __bf16* __restrict__ attn_out) {
  __shared__ __bf16 Ks[64 * 64];
  __shared__ __bf16 Vs[64 * 64];
  __shared__ __bf16 Ps[8][16 * 64];   // per-wave P staging (in-order DS, no barrier)
  const int tid   = threadIdx.x;
  const int lane  = tid & 31;
  const int wv    = tid >> 5;
  const int lrow  = lane & 15;
  const int half8 = (lane >> 4) << 3;
  const int bh    = blockIdx.y;
  const int b     = bh >> 4;
  const int h     = bh & 15;
  const int qbase = blockIdx.x * 128;
  const int qrow0 = qbase + wv * 16;
  const int len   = lengths[b];
  const __bf16* qp = q + (size_t)bh * S_LEN * DK_DIM;
  const __bf16* kp = k + (size_t)bh * S_LEN * DK_DIM;
  const __bf16* vp = v + (size_t)bh * S_LEN * DK_DIM;

  // Q fragments for this wave's 16 rows (A-layout: two contiguous 16B K runs)
  bf16x16 aq[2];
#pragma unroll
  for (int kc = 0; kc < 2; ++kc) {
    const __bf16* rp = qp + (size_t)(qrow0 + lrow) * DK_DIM + kc * 32 + half8;
    aq[kc] = join8(*(const bf16x8*)rp, *(const bf16x8*)(rp + 16));
  }

  f32x8 o[4];
  float mrow[8], lsum[8];
#pragma unroll
  for (int nv = 0; nv < 4; ++nv) o[nv] = f32x8{0.f,0.f,0.f,0.f,0.f,0.f,0.f,0.f};
#pragma unroll
  for (int s = 0; s < 8; ++s) { mrow[s] = NEGF; lsum[s] = 0.f; }

  const int kcount = (qbase + 128 < len) ? (qbase + 128) : len;
  for (int kb0 = 0; kb0 < kcount; kb0 += 64) {
    __syncthreads();
    // ---- K/V tiles: straight copies, async direct-to-LDS ----
#if HAVE_ASYNC_LDS
#pragma unroll
    for (int i = 0; i < 2; ++i) {
      int f = i * 2048 + tid * 8;
      int r = f >> 6, c = f & 63;
      async_copy_b128(&kp[(size_t)(kb0 + r) * DK_DIM + c], &Ks[f]);
      async_copy_b128(&vp[(size_t)(kb0 + r) * DK_DIM + c], &Vs[f]);
    }
    wait_async0();
#else
    {
      bf16x8 tk[2], tv[2];
#pragma unroll
      for (int i = 0; i < 2; ++i) {
        int f = i * 2048 + tid * 8;
        int r = f >> 6, c = f & 63;
        tk[i] = *(const bf16x8*)&kp[(size_t)(kb0 + r) * DK_DIM + c];
        tv[i] = *(const bf16x8*)&vp[(size_t)(kb0 + r) * DK_DIM + c];
      }
#pragma unroll
      for (int i = 0; i < 2; ++i) {
        *(bf16x8*)&Ks[i * 2048 + tid * 8] = tk[i];
        *(bf16x8*)&Vs[i * 2048 + tid * 8] = tv[i];
      }
    }
#endif
    __syncthreads();
    if (kb0 > qrow0 + 15) continue;   // fully masked for this wave (wave-uniform)

    // scores S = Q K^T  (K tile rows are contiguous d-runs -> B fragments)
    f32x8 sc[4];
#pragma unroll
    for (int nt = 0; nt < 4; ++nt) sc[nt] = f32x8{0.f,0.f,0.f,0.f,0.f,0.f,0.f,0.f};
#pragma unroll
    for (int nt = 0; nt < 4; ++nt)
#pragma unroll
      for (int kc = 0; kc < 2; ++kc) {
        const __bf16* rp = &Ks[(nt * 16 + lrow) * 64 + kc * 32 + half8];
        bf16x16 bk = join8(*(const bf16x8*)rp, *(const bf16x8*)(rp + 16));
        sc[nt] = wmma_bf16(aq[kc], bk, sc[nt]);
      }

    // scale + causal/length mask + per-row block max
    float bm[8];
#pragma unroll
    for (int s = 0; s < 8; ++s) bm[s] = NEGF;
#pragma unroll
    for (int nt = 0; nt < 4; ++nt) {
      int kIdx = kb0 + nt * 16 + lrow;
#pragma unroll
      for (int s = 0; s < 8; ++s) {
        int qIdx = qrow0 + s + half8;
        float val = sc[nt][s] * 0.125f;           // 1/sqrt(64)
        bool ok = (kIdx <= qIdx) && (kIdx < len);
        val = ok ? val : NEGF;
        sc[nt][s] = val;
        bm[s] = fmaxf(bm[s], val);
      }
    }
#pragma unroll
    for (int s = 0; s < 8; ++s)
#pragma unroll
      for (int off = 1; off < 16; off <<= 1)
        bm[s] = fmaxf(bm[s], __shfl_xor(bm[s], off, 32));

    float mnew[8], alpha[8], rsum[8];
#pragma unroll
    for (int s = 0; s < 8; ++s) {
      mnew[s]  = fmaxf(mrow[s], bm[s]);
      alpha[s] = __expf(mrow[s] - mnew[s]);
      rsum[s]  = 0.f;
    }
    // P = exp(S - m), stage bf16 to per-wave LDS to re-enter A layout
#pragma unroll
    for (int nt = 0; nt < 4; ++nt)
#pragma unroll
      for (int s = 0; s < 8; ++s) {
        float p = __expf(sc[nt][s] - mnew[s]);
        rsum[s] += p;
        Ps[wv][(s + half8) * 64 + nt * 16 + lrow] = (__bf16)p;
      }
#pragma unroll
    for (int s = 0; s < 8; ++s) {
#pragma unroll
      for (int off = 1; off < 16; off <<= 1) rsum[s] += __shfl_xor(rsum[s], off, 32);
      lsum[s] = lsum[s] * alpha[s] + rsum[s];
      mrow[s] = mnew[s];
    }
#pragma unroll
    for (int nv = 0; nv < 4; ++nv)
#pragma unroll
      for (int s = 0; s < 8; ++s) o[nv][s] *= alpha[s];

    // reload P as A fragments (same-wave DS ops are in-order)
    bf16x16 ap[2];
#pragma unroll
    for (int kc = 0; kc < 2; ++kc) {
      const __bf16* rp = &Ps[wv][lrow * 64 + kc * 32 + half8];
      ap[kc] = join8(*(const bf16x8*)rp, *(const bf16x8*)(rp + 16));
    }
    // O += P V
#pragma unroll
    for (int nv = 0; nv < 4; ++nv)
#pragma unroll
      for (int kc = 0; kc < 2; ++kc) {
        bf16x16 bv;
#pragma unroll
        for (int i = 0; i < 8; ++i) {
          bv[i]     = Vs[(kc * 32 + half8 + i)      * 64 + nv * 16 + lrow];
          bv[i + 8] = Vs[(kc * 32 + half8 + 16 + i) * 64 + nv * 16 + lrow];
        }
        o[nv] = wmma_bf16(ap[kc], bv, o[nv]);
      }
  }

  // write O / l as bf16 into [B,S,H*DV]
#pragma unroll
  for (int s = 0; s < 8; ++s) {
    float inv = 1.0f / lsum[s];
    int qIdx = qrow0 + s + half8;
#pragma unroll
    for (int nv = 0; nv < 4; ++nv) {
      int col = h * DK_DIM + nv * 16 + lrow;
      attn_out[((size_t)b * S_LEN + qIdx) * D_DIM + col] = (__bf16)(o[nv][s] * inv);
    }
  }
}

// ---------------- host launcher ----------------
extern "C" void kernel_launch(void* const* d_in, const int* in_sizes, int n_in,
                              void* d_out, int out_size, void* d_ws, size_t ws_size,
                              hipStream_t stream) {
  (void)in_sizes; (void)n_in; (void)out_size; (void)ws_size;
  const float* x      = (const float*)d_in[0];
  const int*   lens   = (const int*)  d_in[1];
  const float* gamma  = (const float*)d_in[2];
  const float* beta   = (const float*)d_in[3];
  const float* Wq     = (const float*)d_in[4];
  const float* bq     = (const float*)d_in[5];
  const float* Wkv    = (const float*)d_in[6];
  const float* bkv    = (const float*)d_in[7];
  const float* Wo     = (const float*)d_in[8];
  const float* bo     = (const float*)d_in[9];
  float* out = (float*)d_out;

  char* wsp = (char*)d_ws;
  size_t off = 0;
  auto take = [&](size_t bytes) -> void* {
    void* p = wsp + off;
    off += (bytes + 255) & ~(size_t)255;
    return p;
  };
  const size_t ROWS = (size_t)B_NUM * S_LEN;              // 8192
  __bf16* xn    = (__bf16*)take(ROWS * D_DIM * 2);        // 16 MB
  __bf16* Wqkvb = (__bf16*)take((size_t)1024 * 3072 * 2); // 6 MB
  __bf16* Wob   = (__bf16*)take((size_t)1024 * 1024 * 2); // 2 MB
  float*  biasq = (float*) take((size_t)3072 * 4);
  __bf16* qb    = (__bf16*)take(ROWS * D_DIM * 2);        // 16 MB
  __bf16* kb    = (__bf16*)take(ROWS * D_DIM * 2);        // 16 MB
  __bf16* vb    = (__bf16*)take(ROWS * D_DIM * 2);        // 16 MB
  __bf16* attn  = (__bf16*)take(ROWS * D_DIM * 2);        // 16 MB

  mha_cvt_kernel<<<(1024 * 3072) / 256, 256, 0, stream>>>(Wq, Wkv, bq, bkv, Wo,
                                                          Wqkvb, Wob, biasq);
  mha_ln_kernel<<<ROWS / 8, 256, 0, stream>>>(x, gamma, beta, xn);
  mha_gemm_kernel<<<dim3(ROWS / BM, 3072 / BN), 256, 0, stream>>>(
      xn, Wqkvb, 3072, 1024, biasq, nullptr, nullptr, qb, kb, vb, 0);
  mha_attn_kernel<<<dim3(S_LEN / 128, B_NUM * H_NUM), 256, 0, stream>>>(
      qb, kb, vb, lens, attn);
  mha_gemm_kernel<<<dim3(ROWS / BM, 1024 / BN), 256, 0, stream>>>(
      attn, Wob, 1024, 1024, bo, x, out, nullptr, nullptr, nullptr, 1);
}
